// LRURec_26156350832990
// MI455X (gfx1250) — matrix operand order repaired and checked by reading, not compiled, not run on hardware
//
#include <hip/hip_runtime.h>
#include <hip/hip_bf16.h>
#include <math.h>

typedef __bf16 bf16;
typedef __attribute__((ext_vector_type(16))) __bf16 v16bf;
typedef __attribute__((ext_vector_type(8)))  __bf16 v8bf;
typedef __attribute__((ext_vector_type(4)))  __bf16 v4bf;
typedef __attribute__((ext_vector_type(8)))  float  v8f;
typedef __attribute__((ext_vector_type(4)))  float  v4f;

#define B_    512
#define S_    200
#define D_    128
#define NL_   2
#define DH_   256
#define DFF_  512
#define LPAD  256
#define PAD_  (LPAD - S_)          // 56
#define M_TOT (B_ * LPAD)          // 131072

// ---------------------------------------------------------------------------
// Embedding lookup + LayerNorm over D=128, with front padding of zeros.
// ---------------------------------------------------------------------------
__launch_bounds__(128)
__global__ void embed_ln_kernel(const float* __restrict__ token_emb,
                                const float* __restrict__ g,
                                const float* __restrict__ b,
                                const int*   __restrict__ item_seq,
                                float* __restrict__ x) {
    int row = blockIdx.x;              // 0..M_TOT-1
    int t   = row & (LPAD - 1);
    int bb  = row >> 8;
    int d   = threadIdx.x;             // 0..127
    __shared__ float red[128];

    float v = 0.f;
    if (t >= PAD_) {
        int item = item_seq[bb * S_ + (t - PAD_)];
        v = token_emb[(size_t)item * D_ + d];
    }
    red[d] = v;
    __syncthreads();
    for (int s = 64; s > 0; s >>= 1) {
        if (d < s) red[d] += red[d + s];
        __syncthreads();
    }
    float mu = red[0] * (1.0f / D_);
    __syncthreads();
    float dv = v - mu;
    red[d] = dv * dv;
    __syncthreads();
    for (int s = 64; s > 0; s >>= 1) {
        if (d < s) red[d] += red[d + s];
        __syncthreads();
    }
    float inv = rsqrtf(red[0] * (1.0f / D_) + 1e-5f);
    float o = (t >= PAD_) ? (dv * inv * g[d] + b[d]) : 0.f;
    x[(size_t)row * D_ + d] = o;
}

// ---------------------------------------------------------------------------
// Padded mask as float (1.0 / 0.0).
// ---------------------------------------------------------------------------
__global__ void mask_kernel(const int* __restrict__ item_seq,
                            float* __restrict__ maskf) {
    int i = blockIdx.x * blockDim.x + threadIdx.x;
    if (i >= M_TOT) return;
    int t = i & (LPAD - 1);
    int bb = i >> 8;
    maskf[i] = (t >= PAD_ && item_seq[bb * S_ + (t - PAD_)] > 0) ? 1.f : 0.f;
}

// ---------------------------------------------------------------------------
// f32 -> bf16 weight conversion (vectorized: n must be a multiple of 4).
// ---------------------------------------------------------------------------
__global__ void f32_to_bf16_kernel(const float* __restrict__ src,
                                   bf16* __restrict__ dst, int n4) {
    int i = blockIdx.x * blockDim.x + threadIdx.x;
    if (i >= n4) return;
    v4f s = *(const v4f*)(src + i * 4);
    v4bf o;
    o[0] = (bf16)s[0]; o[1] = (bf16)s[1]; o[2] = (bf16)s[2]; o[3] = (bf16)s[3];
    *(v4bf*)(dst + i * 4) = o;
}

// Pack [out_wr | -out_wi] -> bf16 [D_][2*DH_] row-major ([N][K], K=512)
__global__ void pack_wout_kernel(const float* __restrict__ wr,
                                 const float* __restrict__ wi,
                                 bf16* __restrict__ dst) {
    int i = blockIdx.x * blockDim.x + threadIdx.x;   // over 128*512/4
    if (i >= D_ * 2 * DH_ / 4) return;
    int n = (i * 4) >> 9;
    int k = (i * 4) & 511;
    v4f s;
    if (k < DH_) {
        s = *(const v4f*)(wr + n * DH_ + k);
    } else {
        v4f w = *(const v4f*)(wi + n * DH_ + (k - DH_));
        s[0] = -w[0]; s[1] = -w[1]; s[2] = -w[2]; s[3] = -w[3];
    }
    v4bf o;
    o[0] = (bf16)s[0]; o[1] = (bf16)s[1]; o[2] = (bf16)s[2]; o[3] = (bf16)s[3];
    *(v4bf*)(dst + i * 4) = o;
}

// gamma[ch] = exp(params_log[li][2][ch])
__global__ void gamma_kernel(const float* __restrict__ params_log_l,
                             float* __restrict__ gamma) {
    int ch = blockIdx.x * blockDim.x + threadIdx.x;
    if (ch < DH_) gamma[ch] = expf(params_log_l[2 * DH_ + ch]);
}

// ---------------------------------------------------------------------------
// LRU masked log-depth scan in LDS. One block = (batch b, 32 channels).
// ha layout: row = b*LPAD + t, 512 cols: [0..255]=hr, [256..511]=hi.
// Exact reference semantics: level i combines lambda^(j+1) * gate(mid).
// ---------------------------------------------------------------------------
__launch_bounds__(32)
__global__ void lru_scan_kernel(float* __restrict__ ha,
                                const float* __restrict__ maskf,
                                const float* __restrict__ params_log_l) {
    __shared__ float hr[LPAD * 32];   // 32 KB
    __shared__ float hi[LPAD * 32];   // 32 KB
    int bb   = blockIdx.x;
    int cb   = blockIdx.y * 32;
    int lane = threadIdx.x;           // 0..31
    int ch   = cb + lane;

    float nu  = expf(params_log_l[ch]);
    float th  = expf(params_log_l[DH_ + ch]);
    float mag = expf(-nu);
    float lr  = mag * cosf(th);
    float li  = mag * sinf(th);

    size_t base = (size_t)bb * LPAD * (2 * DH_);
    for (int t = 0; t < LPAD; ++t) {
        hr[t * 32 + lane] = ha[base + (size_t)t * (2 * DH_) + ch];
        hi[t * 32 + lane] = ha[base + (size_t)t * (2 * DH_) + DH_ + ch];
    }
    const float* mrow = maskf + (size_t)bb * LPAD;

    for (int lev = 1; lev <= 8; ++lev) {
        int l = 1 << lev, half = l >> 1;
        for (int bs = 0; bs < LPAD; bs += l) {
            int mid = bs + half - 1;
            float gate = mrow[mid];
            float cr = hr[mid * 32 + lane] * gate;
            float ci = hi[mid * 32 + lane] * gate;
            float pr = lr, pi = li;
            for (int j = 0; j < half; ++j) {
                int t = bs + half + j;
                hr[t * 32 + lane] += pr * cr - pi * ci;
                hi[t * 32 + lane] += pr * ci + pi * cr;
                float npr = pr * lr - pi * li;
                float npi = pr * li + pi * lr;
                pr = npr; pi = npi;
            }
        }
    }
    for (int t = 0; t < LPAD; ++t) {
        ha[base + (size_t)t * (2 * DH_) + ch]       = hr[t * 32 + lane];
        ha[base + (size_t)t * (2 * DH_) + DH_ + ch] = hi[t * 32 + lane];
    }
}

// ---------------------------------------------------------------------------
// WMMA bf16 GEMM: C[M,N] = A[M,K] (f32) x Bw[N,K]^T (bf16) with fused epilogue.
// Block tile 64(M) x 128(N), 256 threads = 8 waves (4 M x 2 N), wave tile
// 16x64 = 4 x v_wmma_f32_16x16x32_bf16 accumulators.
// mode 0: C = (acc + bias) * scale                     (input projection)
// mode 1: C = gelu_exact(acc + bias)                   (FFN up)
// mode 2: C = LayerNorm(acc + bias + resid; ln_g,ln_b) (requires N==128)
// ---------------------------------------------------------------------------
__device__ __forceinline__ v16bf load_a_frag(const bf16* As, int m, int lane) {
    int kh = (lane >> 4) << 3;            // 0 or 8
    const bf16* p = As + m * 32;
    v16bf v;
#pragma unroll
    for (int e = 0; e < 8; ++e) v[e] = p[kh + e];
#pragma unroll
    for (int e = 0; e < 8; ++e) v[8 + e] = p[16 + kh + e];
    return v;
}

__device__ __forceinline__ v16bf load_b_frag(const bf16* Bs, int n, int lane) {
    int kh = (lane >> 4) << 3;            // 0 or 8
    const bf16* p = Bs + n * 32;
    v16bf v;
#pragma unroll
    for (int e = 0; e < 8; ++e) v[e] = p[kh + e];
#pragma unroll
    for (int e = 0; e < 8; ++e) v[8 + e] = p[16 + kh + e];
    return v;
}

__launch_bounds__(256)
__global__ void gemm_wmma_kernel(const float* __restrict__ A,
                                 const bf16*  __restrict__ Bw,
                                 const float* __restrict__ bias,
                                 const float* __restrict__ scale,
                                 const float* __restrict__ resid,
                                 const float* __restrict__ ln_g,
                                 const float* __restrict__ ln_b,
                                 float* __restrict__ Cout,
                                 int K, int ldc, int c_off, int mode) {
    __shared__ float smem[64 * 128];      // 32 KB, aliased A/B staging <-> C tile
    __shared__ float s_part[64 * 4 * 2];
    __shared__ float s_mv[64 * 2];
    bf16* As = (bf16*)smem;               // 64 x 32 bf16  (4 KB)
    bf16* Bs = As + 64 * 32;              // 128 x 32 bf16 (8 KB)
    float* Cs = smem;

    int tid   = threadIdx.x;
    int lane  = tid & 31;
    int wid   = tid >> 5;
    int mBase = blockIdx.x * 64;
    int nBase = blockIdx.y * 128;
    int m0 = (wid & 3) * 16;
    int n0 = (wid >> 2) * 64;

    v8f acc[4];
#pragma unroll
    for (int i = 0; i < 4; ++i)
#pragma unroll
        for (int e = 0; e < 8; ++e) acc[i][e] = 0.f;

    for (int k0 = 0; k0 < K; k0 += 32) {
        __syncthreads();
        // A tile: 64x32 f32 -> bf16, vectorized (512 float4 / 256 threads)
#pragma unroll
        for (int it = 0; it < 2; ++it) {
            int idx = tid + it * 256;
            int r = idx >> 3, c4 = (idx & 7) << 2;
            const float* ap = A + (size_t)(mBase + r) * K + k0 + c4;
            v4f av = *(const v4f*)ap;
            v4bf o;
            o[0] = (bf16)av[0]; o[1] = (bf16)av[1];
            o[2] = (bf16)av[2]; o[3] = (bf16)av[3];
            *(v4bf*)(As + r * 32 + c4) = o;
            if (k0 + 32 < K) __builtin_prefetch(ap + 32, 0, 1);
        }
        // B tile: 128x32 bf16 copy, vectorized (512 x 16B / 256 threads)
#pragma unroll
        for (int it = 0; it < 2; ++it) {
            int idx = tid + it * 256;
            int r = idx >> 2, c8 = (idx & 3) << 3;
            *(v8bf*)(Bs + r * 32 + c8) =
                *(const v8bf*)(Bw + (size_t)(nBase + r) * K + k0 + c8);
        }
        __syncthreads();
        v16bf af = load_a_frag(As, m0 + (lane & 15), lane);
#pragma unroll
        for (int nt = 0; nt < 4; ++nt) {
            v16bf bfr = load_b_frag(Bs, n0 + nt * 16 + (lane & 15), lane);
            acc[nt] = __builtin_amdgcn_wmma_f32_16x16x32_bf16(
                false, af, false, bfr, (short)0, acc[nt], false, false);
        }
    }
    __syncthreads();

    // Scatter accumulators into the C tile (ISA 16x16 f32 C/D layout).
#pragma unroll
    for (int nt = 0; nt < 4; ++nt) {
        int col   = n0 + nt * 16 + (lane & 15);
        int rbase = m0 + ((lane >> 4) << 3);
#pragma unroll
        for (int v = 0; v < 8; ++v)
            Cs[(rbase + v) * 128 + col] = acc[nt][v];
    }
    __syncthreads();

    if (mode == 0) {
        for (int idx = tid; idx < 2048; idx += 256) {
            int r = idx >> 5, c4 = (idx & 31) << 2;
            int gn = nBase + c4;
            v4f v = *(v4f*)(Cs + r * 128 + c4);
#pragma unroll
            for (int e = 0; e < 4; ++e) v[e] = (v[e] + bias[gn + e]) * scale[gn + e];
            *(v4f*)(Cout + (size_t)(mBase + r) * ldc + c_off + gn) = v;
        }
    } else if (mode == 1) {
        for (int idx = tid; idx < 2048; idx += 256) {
            int r = idx >> 5, c4 = (idx & 31) << 2;
            int gn = nBase + c4;
            v4f v = *(v4f*)(Cs + r * 128 + c4);
#pragma unroll
            for (int e = 0; e < 4; ++e) {
                float t = v[e] + bias[gn + e];
                v[e] = 0.5f * t * (1.0f + erff(t * 0.70710678118654752f));
            }
            *(v4f*)(Cout + (size_t)(mBase + r) * ldc + c_off + gn) = v;
        }
    } else {
        // mode 2: bias + residual + LayerNorm over 128 cols (nBase == 0)
        for (int idx = tid; idx < 2048; idx += 256) {
            int r = idx >> 5, c4 = (idx & 31) << 2;
            v4f v = *(v4f*)(Cs + r * 128 + c4);
            v4f rv = *(const v4f*)(resid + (size_t)(mBase + r) * 128 + c4);
#pragma unroll
            for (int e = 0; e < 4; ++e) v[e] += bias[c4 + e] + rv[e];
            *(v4f*)(Cs + r * 128 + c4) = v;
        }
        __syncthreads();
        {
            int r = tid >> 2, q = tid & 3;
            float s = 0.f, s2 = 0.f;
            const float* rowp = Cs + r * 128 + q * 32;
            for (int c = 0; c < 32; ++c) { float v = rowp[c]; s += v; s2 += v * v; }
            s_part[(r * 4 + q) * 2 + 0] = s;
            s_part[(r * 4 + q) * 2 + 1] = s2;
        }
        __syncthreads();
        if (tid < 64) {
            float s = 0.f, s2 = 0.f;
            for (int q = 0; q < 4; ++q) {
                s  += s_part[(tid * 4 + q) * 2 + 0];
                s2 += s_part[(tid * 4 + q) * 2 + 1];
            }
            float mu  = s * (1.0f / 128.f);
            float var = s2 * (1.0f / 128.f) - mu * mu;
            s_mv[tid * 2 + 0] = mu;
            s_mv[tid * 2 + 1] = rsqrtf(var + 1e-5f);
        }
        __syncthreads();
        for (int idx = tid; idx < 2048; idx += 256) {
            int r = idx >> 5, c4 = (idx & 31) << 2;
            float mu = s_mv[r * 2], inv = s_mv[r * 2 + 1];
            v4f v = *(v4f*)(Cs + r * 128 + c4);
#pragma unroll
            for (int e = 0; e < 4; ++e)
                v[e] = (v[e] - mu) * inv * ln_g[c4 + e] + ln_b[c4 + e];
            *(v4f*)(Cout + (size_t)(mBase + r) * ldc + c_off + c4) = v;
        }
    }
}

// ---------------------------------------------------------------------------
// Final gather: out[b] = x[b, PAD_ + len[b] - 1]
// ---------------------------------------------------------------------------
__global__ void gather_kernel(const float* __restrict__ x,
                              const int* __restrict__ seq_len,
                              float* __restrict__ out) {
    int bb = blockIdx.x;
    int d  = threadIdx.x;
    int t  = PAD_ + seq_len[bb] - 1;
    out[(size_t)bb * D_ + d] = x[((size_t)bb * LPAD + t) * D_ + d];
}

// ---------------------------------------------------------------------------
extern "C" void kernel_launch(void* const* d_in, const int* in_sizes, int n_in,
                              void* d_out, int out_size, void* d_ws, size_t ws_size,
                              hipStream_t stream) {
    (void)in_sizes; (void)n_in; (void)out_size; (void)ws_size;

    const float* token_emb  = (const float*)d_in[0];
    const float* emb_ln_g   = (const float*)d_in[1];
    const float* emb_ln_b   = (const float*)d_in[2];
    const float* params_log = (const float*)d_in[3];
    const float* in_wr      = (const float*)d_in[4];
    const float* in_wi      = (const float*)d_in[5];
    const float* in_br      = (const float*)d_in[6];
    const float* in_bi      = (const float*)d_in[7];
    const float* out_wr     = (const float*)d_in[8];
    const float* out_wi     = (const float*)d_in[9];
    const float* out_br     = (const float*)d_in[10];
    const float* out_bi     = (const float*)d_in[11]; (void)out_bi; // imag of y discarded
    const float* lru_ln_g   = (const float*)d_in[12];
    const float* lru_ln_b   = (const float*)d_in[13];
    const float* w1         = (const float*)d_in[14];
    const float* b1         = (const float*)d_in[15];
    const float* w2         = (const float*)d_in[16];
    const float* b2         = (const float*)d_in[17];
    const float* ffn_ln_g   = (const float*)d_in[18];
    const float* ffn_ln_b   = (const float*)d_in[19];
    const int*   item_seq   = (const int*)d_in[20];
    const int*   item_len   = (const int*)d_in[21];
    float* out = (float*)d_out;

    char* ws = (char*)d_ws;
    size_t off = 0;
    auto alloc = [&](size_t bytes) -> void* {
        off = (off + 255) & ~(size_t)255;
        void* p = ws + off;
        off += bytes;
        return p;
    };
    float* x0     = (float*)alloc((size_t)M_TOT * D_ * 4);
    float* x1     = (float*)alloc((size_t)M_TOT * D_ * 4);
    float* ha     = (float*)alloc((size_t)M_TOT * 2 * DH_ * 4); // also reused as FFN hidden
    float* maskf  = (float*)alloc((size_t)M_TOT * 4);
    float* gamma  = (float*)alloc(DH_ * 4);
    bf16*  wb_in_r = (bf16*)alloc((size_t)DH_ * D_ * 2);
    bf16*  wb_in_i = (bf16*)alloc((size_t)DH_ * D_ * 2);
    bf16*  wb_out  = (bf16*)alloc((size_t)D_ * 2 * DH_ * 2);
    bf16*  wb1     = (bf16*)alloc((size_t)DFF_ * D_ * 2);
    bf16*  wb2     = (bf16*)alloc((size_t)D_ * DFF_ * 2);

    embed_ln_kernel<<<M_TOT, 128, 0, stream>>>(token_emb, emb_ln_g, emb_ln_b, item_seq, x0);
    mask_kernel<<<(M_TOT + 255) / 256, 256, 0, stream>>>(item_seq, maskf);

    float* xc = x0;   // layer input / residual, also final output buffer
    float* xn = x1;

    for (int li = 0; li < NL_; ++li) {
        const float* pl = params_log + (size_t)li * 3 * DH_;

        f32_to_bf16_kernel<<<(DH_ * D_ / 4 + 255) / 256, 256, 0, stream>>>(
            in_wr + (size_t)li * DH_ * D_, wb_in_r, DH_ * D_ / 4);
        f32_to_bf16_kernel<<<(DH_ * D_ / 4 + 255) / 256, 256, 0, stream>>>(
            in_wi + (size_t)li * DH_ * D_, wb_in_i, DH_ * D_ / 4);
        pack_wout_kernel<<<(D_ * 2 * DH_ / 4 + 255) / 256, 256, 0, stream>>>(
            out_wr + (size_t)li * D_ * DH_, out_wi + (size_t)li * D_ * DH_, wb_out);
        f32_to_bf16_kernel<<<(DFF_ * D_ / 4 + 255) / 256, 256, 0, stream>>>(
            w1 + (size_t)li * DFF_ * D_, wb1, DFF_ * D_ / 4);
        f32_to_bf16_kernel<<<(D_ * DFF_ / 4 + 255) / 256, 256, 0, stream>>>(
            w2 + (size_t)li * D_ * DFF_, wb2, D_ * DFF_ / 4);
        gamma_kernel<<<1, 256, 0, stream>>>(pl, gamma);

        // Complex input projection: hr = (x @ in_wr^T + in_br)*gamma ; hi likewise.
        dim3 gIn(M_TOT / 64, DH_ / 128);
        gemm_wmma_kernel<<<gIn, 256, 0, stream>>>(
            xc, wb_in_r, in_br + (size_t)li * DH_, gamma,
            nullptr, nullptr, nullptr, ha, D_, 2 * DH_, 0, 0);
        gemm_wmma_kernel<<<gIn, 256, 0, stream>>>(
            xc, wb_in_i, in_bi + (size_t)li * DH_, gamma,
            nullptr, nullptr, nullptr, ha, D_, 2 * DH_, DH_, 0);

        // Masked LRU scan in LDS.
        lru_scan_kernel<<<dim3(B_, DH_ / 32), 32, 0, stream>>>(ha, maskf, pl);

        // Output projection (K=512, packed [wr | -wi]) + out_br + residual + LN.
        gemm_wmma_kernel<<<dim3(M_TOT / 64, 1), 256, 0, stream>>>(
            ha, wb_out, out_br + (size_t)li * D_, nullptr,
            xc, lru_ln_g + (size_t)li * D_, lru_ln_b + (size_t)li * D_,
            xn, 2 * DH_, D_, 0, 2);

        // FFN up: gelu(x @ w1^T + b1) -> hff (reuse ha).
        gemm_wmma_kernel<<<dim3(M_TOT / 64, DFF_ / 128), 256, 0, stream>>>(
            xn, wb1, b1 + (size_t)li * DFF_, nullptr,
            nullptr, nullptr, nullptr, ha, D_, DFF_, 0, 1);

        // FFN down: hff @ w2^T + b2 + residual + LN -> xc (next layer input).
        gemm_wmma_kernel<<<dim3(M_TOT / 64, 1), 256, 0, stream>>>(
            ha, wb2, b2 + (size_t)li * D_, nullptr,
            xn, ffn_ln_g + (size_t)li * D_, ffn_ln_b + (size_t)li * D_,
            xc, DFF_, D_, 0, 2);
    }

    gather_kernel<<<B_, D_, 0, stream>>>(xc, item_len, out);
}
